// PulseForCausalLM_77000173682700
// MI455X (gfx1250) — matrix operand order repaired and checked by reading, not compiled, not run on hardware
//
#include <hip/hip_runtime.h>
#include <hip/hip_bf16.h>
#include <math.h>

// ---------------- problem constants (from reference) ----------------
#define VOCAB   50257
#define HDIM    1024
#define SEQ     2048
#define BATCH   2
#define ROWS    (BATCH * SEQ)      // 4096 flattened (b,s) rows
#define SOFT_CAP 30.0f
#define ZLOSS    1.0e-4f

// ---------------- tiling (TDM path) ----------------
#define BM 64          // rows per block
#define BN 256         // vocab cols per block
#define BK 32          // k-step (matches wmma 16x16x32 bf16)
#define APITCH 40      // LDS pitch in bf16 elems: 64B row + 16B pad = 80B (TDM pad codes 3/3)
#define BPITCH 40
// ---------------- tiling (fallback path, round-1 kernel) ----------------
#define FBN 128

typedef __attribute__((ext_vector_type(16))) __bf16 v16bf;
typedef __attribute__((ext_vector_type(8)))  __bf16 v8bf;
typedef __attribute__((ext_vector_type(8)))  float  v8f;
typedef __attribute__((ext_vector_type(4)))  unsigned int v4u;
typedef __attribute__((ext_vector_type(4)))  int v4i;
typedef __attribute__((ext_vector_type(8)))  int v8i;

// ---------------- small helpers ----------------
// RNE f32->bf16 (fallback path only)
__device__ __forceinline__ unsigned short f2bf(float f) {
  unsigned u = __builtin_bit_cast(unsigned, f);
  unsigned r = 0x7FFFu + ((u >> 16) & 1u);
  return (unsigned short)((u + r) >> 16);
}
__device__ __forceinline__ unsigned pack2(float lo, float hi) {
  return (unsigned)f2bf(lo) | ((unsigned)f2bf(hi) << 16);
}

struct Frag32 { uint4 lo, hi; };
__device__ __forceinline__ v16bf load_frag(const unsigned short* p0,
                                           const unsigned short* p1) {
  Frag32 t;
  t.lo = *(const uint4*)p0;   // 16B LDS read
  t.hi = *(const uint4*)p1;   // 16B LDS read
  return __builtin_bit_cast(v16bf, t);
}

// LDS byte offset of a __shared__ object (ptrtoint of addrspace(3) pointer)
typedef __attribute__((address_space(3))) const void as3void;
__device__ __forceinline__ unsigned ldsOffset(const void* p) {
  return (unsigned)(unsigned long long)(as3void*)p;
}

// ---------------- TDM 2-D tile load descriptor (D#) ----------------
// tile0 elems along contiguous dim (k), tile1 rows; dims are *remaining* extents
// from the tile start (OOB reads zero-fill). data_size=2B. LDS padding: every
// 16 DWORDs (64B = one 32-elem bf16 row) insert 4 DWORDs (16B) -> 80B pitch.
__device__ __forceinline__ void tdm_load_tile_2d(
    unsigned ldsOff, const void* gsrc,
    unsigned dim0, unsigned dim1, unsigned stride0,
    unsigned tile0, unsigned tile1)
{
  unsigned long long ga = (unsigned long long)gsrc;
  v4u g0;
  g0[0] = 1u;                                   // count=1 (valid user D#)
  g0[1] = ldsOff;                               // lds_addr (bytes)
  g0[2] = (unsigned)ga;                         // global_addr[31:0]
  g0[3] = (unsigned)((ga >> 32) & 0x01FFFFFFull) | (2u << 30); // addr[56:32] | type=2
  v8i g1;
  g1[0] = (int)((1u << 16) | (1u << 20) | (3u << 22) | (3u << 25)); // 2B elems, pad_en, 16DW interval, 4DW pad
  g1[1] = (int)((dim0 & 0xFFFFu) << 16);                 // tensor_dim0[15:0]
  g1[2] = (int)((dim0 >> 16) | ((dim1 & 0xFFFFu) << 16)); // dim0[31:16] | dim1[15:0]
  g1[3] = (int)((dim1 >> 16) | (tile0 << 16));            // dim1[31:16] | tile_dim0
  g1[4] = (int)(tile1 & 0xFFFFu);                         // tile_dim1 (tile_dim2=0)
  g1[5] = (int)stride0;                                   // tensor_dim0_stride[31:0]
  g1[6] = 0;
  g1[7] = 0;
  v4i gz = {0, 0, 0, 0};
#if __clang_major__ >= 23
  v8i gz8 = {0, 0, 0, 0, 0, 0, 0, 0};
  __builtin_amdgcn_tensor_load_to_lds(g0, g1, gz, gz, gz8, 0);  // clang-23 arity
#else
  __builtin_amdgcn_tensor_load_to_lds(g0, g1, gz, gz, 0);       // ROCm 7.2 / clang-22 arity
#endif
}

// ---------------- kernel 0: shifted labels + zero accumulators ----------------
__global__ void prep_kernel(const long long* __restrict__ labels,
                            int* __restrict__ lab,
                            float* __restrict__ gSumExp,
                            float* __restrict__ gPicked) {
  int i = blockIdx.x * blockDim.x + threadIdx.x;
  if (i >= ROWS) return;
  int b = i / SEQ, s = i % SEQ;
  long long L = (s < SEQ - 1) ? labels[(long long)b * SEQ + s + 1] : (long long)-100;
  lab[i] = (L == (long long)-100) ? -1 : (int)L;
  gSumExp[i] = 0.0f;
  gPicked[i] = 0.0f;
}

// ---------------- kernel 0b: f32 -> bf16 cast (packed hardware converts) ----------------
typedef __attribute__((ext_vector_type(8))) float v8fv;
__global__ __launch_bounds__(256) void cast_bf16_kernel(const float* __restrict__ src,
                                                        __bf16* __restrict__ dst,
                                                        long long n8) {
  long long i = (long long)blockIdx.x * blockDim.x + threadIdx.x;
  long long stride = (long long)gridDim.x * blockDim.x;
  for (; i < n8; i += stride) {
    const float4* p = (const float4*)(src + i * 8);
    float4 f0 = p[0], f1 = p[1];
    v8fv f = {f0.x, f0.y, f0.z, f0.w, f1.x, f1.y, f1.z, f1.w};
    v8bf b = __builtin_convertvector(f, v8bf);
    *(uint4*)(dst + i * 8) = __builtin_bit_cast(uint4, b);
  }
}

// ---------------- kernel 1: TDM double-buffered bf16 WMMA GEMM + softcap/exp ----------------
__global__ __launch_bounds__(256) void ce_gemm_tdm(
    const __bf16* __restrict__ hb,   // [ROWS][HDIM] bf16
    const __bf16* __restrict__ wb,   // [VOCAB][HDIM] bf16
    const int*   __restrict__ lab,
    float* __restrict__ gSumExp,
    float* __restrict__ gPicked)
{
  __shared__ __align__(16) unsigned short ldsA[2][BM * APITCH];
  __shared__ __align__(16) unsigned short ldsB[2][BN * BPITCH];
  __shared__ float sExp[BM];
  __shared__ int   sLab[BM];

  const int t     = threadIdx.x;
  const int mbase = blockIdx.y * BM;
  const int nbase = blockIdx.x * BN;

  if (t < BM) {
    sExp[t] = 0.0f;
    sLab[t] = lab[mbase + t];
  }

  const int wid  = t >> 5;
  const int lane = t & 31;
  const int hl   = lane >> 4;
  const int ll   = lane & 15;
  const int wm   = wid >> 2;      // 0..1 : 32-row strip
  const int wn   = wid & 3;       // 0..3 : 64-col strip

  v8f zero = {};
  v8f c[2][4];
#pragma unroll
  for (int i = 0; i < 2; ++i)
#pragma unroll
    for (int j = 0; j < 4; ++j) c[i][j] = zero;

  const __bf16* aSrc = hb + (size_t)mbase * HDIM;
  const __bf16* bSrc = wb + (size_t)nbase * HDIM;

  // prologue: TDM fill of buffer 0 (wave 0 issues; EXEC-independent DMA)
  if (wid == 0) {
    tdm_load_tile_2d(ldsOffset(&ldsA[0][0]), aSrc, HDIM, (unsigned)(ROWS - mbase),
                     HDIM, BK, BM);
    tdm_load_tile_2d(ldsOffset(&ldsB[0][0]), bSrc, HDIM, (unsigned)(VOCAB - nbase),
                     HDIM, BK, BN);
  }

  const int am0 = wm * 32 + ll;
  const int am1 = wm * 32 + 16 + ll;

#pragma unroll 1
  for (int k0 = 0, it = 0; k0 < HDIM; k0 += BK, ++it) {
    const int buf = it & 1;
    if (wid == 0) __builtin_amdgcn_s_wait_tensorcnt(0);  // current slice landed
    __syncthreads();
    if (wid == 0 && k0 + BK < HDIM) {                    // prefetch next slice via TDM
      const int nx = buf ^ 1;
      tdm_load_tile_2d(ldsOffset(&ldsA[nx][0]), aSrc + (k0 + BK),
                       (unsigned)(HDIM - (k0 + BK)), (unsigned)(ROWS - mbase),
                       HDIM, BK, BM);
      tdm_load_tile_2d(ldsOffset(&ldsB[nx][0]), bSrc + (k0 + BK),
                       (unsigned)(HDIM - (k0 + BK)), (unsigned)(VOCAB - nbase),
                       HDIM, BK, BN);
    }

    const unsigned short* A  = ldsA[buf];
    const unsigned short* Bt = ldsB[buf];
    // A 16x32 bf16 frag: lane=row; lo-half k=0..15, hi-half k=16..31
    v16bf a0 = load_frag(&A[am0 * APITCH + hl * 16], &A[am0 * APITCH + hl * 16 + 8]);
    v16bf a1 = load_frag(&A[am1 * APITCH + hl * 16], &A[am1 * APITCH + hl * 16 + 8]);
#pragma unroll
    for (int j = 0; j < 4; ++j) {
      const int bn = wn * 64 + j * 16 + ll;
      // B 32x16 bf16 frag: lane=col; lo-half k=0..15, hi-half k=16..31 (k-contig in LDS[n][k])
      v16bf b = load_frag(&Bt[bn * BPITCH + hl * 16], &Bt[bn * BPITCH + hl * 16 + 8]);
      c[0][j] = __builtin_amdgcn_wmma_f32_16x16x32_bf16(false, a0, false, b, (short)0, c[0][j], false, false);
      c[1][j] = __builtin_amdgcn_wmma_f32_16x16x32_bf16(false, a1, false, b, (short)0, c[1][j], false, false);
    }
  }

  // -------- epilogue: softcap -> exp partial sums (ds_add_f32) + label pick --------
  auto emit = [&](v8f cacc, int mloc_base, int nloc_base) {
    const int nglob = nbase + nloc_base + ll;
#pragma unroll
    for (int r = 0; r < 8; ++r) {
      const int mloc = mloc_base + hl * 8 + r;
      float capped = SOFT_CAP * tanhf(cacc[r] * (1.0f / SOFT_CAP));
      if (nglob < VOCAB) {
        atomicAdd(&sExp[mloc], __expf(capped));
        if (sLab[mloc] == nglob) gPicked[mbase + mloc] = capped;
      }
    }
  };
#pragma unroll
  for (int i = 0; i < 2; ++i)
#pragma unroll
    for (int j = 0; j < 4; ++j) emit(c[i][j], wm * 32 + i * 16, wn * 64 + j * 16);

  __syncthreads();
  if (t < BM) atomicAdd(&gSumExp[mbase + t], sExp[t]);
}

// ---------------- kernel 1-alt: fallback (on-the-fly f32->bf16, small ws) ----------------
__global__ __launch_bounds__(256) void ce_gemm_fallback(
    const float* __restrict__ hidden,
    const float* __restrict__ weight,
    const int*   __restrict__ lab,
    float* __restrict__ gSumExp,
    float* __restrict__ gPicked)
{
  __shared__ __align__(16) unsigned short ldsA[BM * APITCH];
  __shared__ __align__(16) unsigned short ldsB[FBN * BPITCH];
  __shared__ float sExp[BM];
  __shared__ int   sLab[BM];

  const int t     = threadIdx.x;
  const int mbase = blockIdx.y * BM;
  const int nbase = blockIdx.x * FBN;

  if (t < BM) { sExp[t] = 0.0f; sLab[t] = lab[mbase + t]; }

  const int wid  = t >> 5;
  const int lane = t & 31;
  const int hl   = lane >> 4;
  const int ll   = lane & 15;
  const int wm   = wid >> 2;
  const int wn   = wid & 3;

  v8f c00 = {}; v8f c01 = {}; v8f c10 = {}; v8f c11 = {};

  const int arow  = t >> 2;
  const int akoff = (t & 3) * 8;
  const int brow  = t >> 1;
  const int bkoff = (t & 1) * 16;
  const long long aBase = (long long)(mbase + arow) * HDIM + akoff;
  const int vglob = nbase + brow;
  const long long bBase = (long long)vglob * HDIM + bkoff;

#pragma unroll 1
  for (int k0 = 0; k0 < HDIM; k0 += BK) {
    __syncthreads();
    if (vglob < VOCAB && k0 + BK < HDIM) {
      __builtin_prefetch(weight + bBase + k0 + BK, 0, 1);
      __builtin_prefetch(hidden + aBase + k0 + BK, 0, 1);
    }
    {
      const float4* p = (const float4*)(hidden + aBase + k0);
      float4 f0 = p[0], f1 = p[1];
      uint4 w;
      w.x = pack2(f0.x, f0.y); w.y = pack2(f0.z, f0.w);
      w.z = pack2(f1.x, f1.y); w.w = pack2(f1.z, f1.w);
      *(uint4*)&ldsA[arow * APITCH + akoff] = w;
    }
    {
      uint4 w0 = {0u,0u,0u,0u}, w1 = {0u,0u,0u,0u};
      if (vglob < VOCAB) {
        const float4* p = (const float4*)(weight + bBase + k0);
        float4 f0 = p[0], f1 = p[1], f2 = p[2], f3 = p[3];
        w0.x = pack2(f0.x, f0.y); w0.y = pack2(f0.z, f0.w);
        w0.z = pack2(f1.x, f1.y); w0.w = pack2(f1.z, f1.w);
        w1.x = pack2(f2.x, f2.y); w1.y = pack2(f2.z, f2.w);
        w1.z = pack2(f3.x, f3.y); w1.w = pack2(f3.z, f3.w);
      }
      *(uint4*)&ldsB[brow * BPITCH + bkoff]     = w0;
      *(uint4*)&ldsB[brow * BPITCH + bkoff + 8] = w1;
    }
    __syncthreads();

    const int am0 = wm * 32 + ll;
    const int am1 = wm * 32 + 16 + ll;
    v16bf a0 = load_frag(&ldsA[am0 * APITCH + hl * 16], &ldsA[am0 * APITCH + hl * 16 + 8]);
    v16bf a1 = load_frag(&ldsA[am1 * APITCH + hl * 16], &ldsA[am1 * APITCH + hl * 16 + 8]);
    const int bn0 = wn * 32 + ll;
    const int bn1 = wn * 32 + 16 + ll;
    v16bf b0 = load_frag(&ldsB[bn0 * BPITCH + hl * 16], &ldsB[bn0 * BPITCH + hl * 16 + 8]);
    v16bf b1 = load_frag(&ldsB[bn1 * BPITCH + hl * 16], &ldsB[bn1 * BPITCH + hl * 16 + 8]);

    c00 = __builtin_amdgcn_wmma_f32_16x16x32_bf16(false, a0, false, b0, (short)0, c00, false, false);
    c01 = __builtin_amdgcn_wmma_f32_16x16x32_bf16(false, a0, false, b1, (short)0, c01, false, false);
    c10 = __builtin_amdgcn_wmma_f32_16x16x32_bf16(false, a1, false, b0, (short)0, c10, false, false);
    c11 = __builtin_amdgcn_wmma_f32_16x16x32_bf16(false, a1, false, b1, (short)0, c11, false, false);
  }

  auto emit = [&](v8f cacc, int mloc_base, int nloc_base) {
    const int nglob = nbase + nloc_base + ll;
#pragma unroll
    for (int r = 0; r < 8; ++r) {
      const int mloc = mloc_base + hl * 8 + r;
      float capped = SOFT_CAP * tanhf(cacc[r] * (1.0f / SOFT_CAP));
      if (nglob < VOCAB) {
        atomicAdd(&sExp[mloc], __expf(capped));
        if (sLab[mloc] == nglob) gPicked[mbase + mloc] = capped;
      }
    }
  };
  emit(c00, wm * 32,      wn * 32);
  emit(c01, wm * 32,      wn * 32 + 16);
  emit(c10, wm * 32 + 16, wn * 32);
  emit(c11, wm * 32 + 16, wn * 32 + 16);

  __syncthreads();
  if (t < BM) atomicAdd(&gSumExp[mbase + t], sExp[t]);
}

// ---------------- kernel 2: final reduction to scalar loss ----------------
__global__ __launch_bounds__(256) void finalize_kernel(
    const int* __restrict__ lab,
    const float* __restrict__ gSumExp,
    const float* __restrict__ gPicked,
    float* __restrict__ out)
{
  __shared__ float sCe[256];
  __shared__ float sZ[256];
  __shared__ int   sN[256];
  float ce = 0.0f, z = 0.0f; int n = 0;
  for (int i = threadIdx.x; i < ROWS; i += 256) {
    if (lab[i] >= 0) {
      float lz = __logf(gSumExp[i]);   // softcap bounds logits: no max-sub needed
      ce += lz - gPicked[i];
      z  += lz * lz;
      n  += 1;
    }
  }
  sCe[threadIdx.x] = ce; sZ[threadIdx.x] = z; sN[threadIdx.x] = n;
  __syncthreads();
  for (int off = 128; off > 0; off >>= 1) {
    if (threadIdx.x < off) {
      sCe[threadIdx.x] += sCe[threadIdx.x + off];
      sZ[threadIdx.x]  += sZ[threadIdx.x + off];
      sN[threadIdx.x]  += sN[threadIdx.x + off];
    }
    __syncthreads();
  }
  if (threadIdx.x == 0) {
    float nv = (float)(sN[0] > 0 ? sN[0] : 1);
    out[0] = sCe[0] / nv + ZLOSS * sZ[0] / nv;
  }
}

// ---------------- host-side launcher ----------------
#define OFF_SUM   ((size_t)0)
#define OFF_PICK  ((size_t)16384)
#define OFF_LAB   ((size_t)32768)
#define OFF_WBF   ((size_t)49152)
#define WELEMS    ((size_t)VOCAB * HDIM)                  // 51,463,168
#define HELEMS    ((size_t)ROWS * HDIM)                   // 4,194,304
#define OFF_HBF   (OFF_WBF + WELEMS * 2)
#define WS_NEED   (OFF_HBF + HELEMS * 2)                  // ~111.4 MB

extern "C" void kernel_launch(void* const* d_in, const int* in_sizes, int n_in,
                              void* d_out, int out_size, void* d_ws, size_t ws_size,
                              hipStream_t stream) {
  const float*     hidden = (const float*)d_in[0];      // (2,2048,1024) f32
  const float*     weight = (const float*)d_in[1];      // (50257,1024) f32
  const long long* labels = (const long long*)d_in[2];  // (2,2048) i64
  float* out = (float*)d_out;

  char* ws = (char*)d_ws;
  float*  gSumExp = (float*)(ws + OFF_SUM);
  float*  gPicked = (float*)(ws + OFF_PICK);
  int*    gLab    = (int*)  (ws + OFF_LAB);
  __bf16* wbf     = (__bf16*)(ws + OFF_WBF);
  __bf16* hbf     = (__bf16*)(ws + OFF_HBF);

  prep_kernel<<<(ROWS + 255) / 256, 256, 0, stream>>>(labels, gLab, gSumExp, gPicked);

  if (ws_size >= WS_NEED) {
    // Pre-cast once (HBM-bound, ~15us); bf16 copies then live in the 192MB L2.
    cast_bf16_kernel<<<2048, 256, 0, stream>>>(weight, wbf, (long long)(WELEMS / 8));
    cast_bf16_kernel<<<512,  256, 0, stream>>>(hidden, hbf, (long long)(HELEMS / 8));
    dim3 grid((VOCAB + BN - 1) / BN, ROWS / BM);   // 197 x 64
    ce_gemm_tdm<<<grid, 256, 0, stream>>>(hbf, wbf, gLab, gSumExp, gPicked);
  } else {
    dim3 grid((VOCAB + FBN - 1) / FBN, ROWS / BM); // 393 x 64
    ce_gemm_fallback<<<grid, 256, 0, stream>>>(hidden, weight, gLab, gSumExp, gPicked);
  }

  finalize_kernel<<<1, 256, 0, stream>>>(gLab, gSumExp, gPicked, out);
}
